// STConv_58609123721605
// MI455X (gfx1250) — matrix operand order repaired and checked by reading, not compiled, not run on hardware
//
#include <hip/hip_runtime.h>
#include <math.h>

// ---------------------------------------------------------------------------
// CDNA5 (gfx1250) WMMA helpers: bf16 16x16x32, f32 accumulate
// ---------------------------------------------------------------------------
typedef __attribute__((ext_vector_type(16))) __bf16    v16bf;
typedef __attribute__((ext_vector_type(8)))  float     v8f;
typedef __attribute__((ext_vector_type(4)))  unsigned  u32x4;
typedef unsigned short bf16_t;   // storage type for bf16 buffers

union FragLd { v16bf v; u32x4 q[2]; };

__device__ __forceinline__ unsigned short f2bf(float f) {
  union { float f; unsigned u; } x; x.f = f;
  unsigned r = x.u + 0x7FFFu + ((x.u >> 16) & 1u);   // round-to-nearest-even
  return (unsigned short)(r >> 16);
}

// A-matrix 16x32 (MxK) bf16, contiguous k order (k = 16h + s), same as packed B.
// Per lane: u[0..15] = row[kb+0..15]  (kb = 16 for hi lanes)
__device__ __forceinline__ v16bf load_a_bf(const bf16_t* __restrict__ src, int ldr, int lane) {
  const int m  = lane & 15;
  const int kb = (lane & 16) ? 16 : 0;
  const bf16_t* r = src + m * ldr + kb;
  FragLd a;
  a.q[0] = *(const u32x4*)(r);
  a.q[1] = *(const u32x4*)(r + 8);
  return a.v;
}

// B-matrix 32x16 (KxN) bf16, pre-packed tile: [lane][16] contiguous
__device__ __forceinline__ v16bf load_b_pk(const bf16_t* __restrict__ tile, int lane) {
  const u32x4* p = (const u32x4*)(tile + lane * 16);
  FragLd b;
  b.q[0] = p[0];
  b.q[1] = p[1];
  return b.v;
}

#define WMMA_BF16(A, B, C) \
  __builtin_amdgcn_wmma_f32_16x16x32_bf16(false, (A), false, (B), (short)0, (C), false, false)

// ---------------------------------------------------------------------------
// Problem constants
// ---------------------------------------------------------------------------
#define NB   2
#define TT   16
#define NN   2048
#define CIN  32
#define CH   64
#define COUT 64
#define KS   3
#define NE   32768
#define T1D  14            // T - 2
#define T2D  12            // T - 4
#define TILE 512           // bf16 elements per packed B tile (32 lanes x 16)
#define MT   4             // 16-row M-tiles per block (64 rows/block)

// ---------------------------------------------------------------------------
// Weight packing: fragment-ordered bf16 tiles, tile = [kk][ntile][lane][16]
// B element (k_local, n): k_local = 16*(lane>=16) + j ; n = ntile*16 + (lane&15)
// ---------------------------------------------------------------------------
__global__ void k_pack_tc1(const float* __restrict__ w, bf16_t* __restrict__ dst) {
  const int tid = blockIdx.x * blockDim.x + threadIdx.x;   // 3*4*512 = 6144
  if (tid >= 3 * 4 * TILE) return;
  const int j    = tid & 15;
  const int lane = (tid >> 4) & 31;
  const int nt   = (tid >> 9) & 3;
  const int kk   = tid >> 11;                              // temporal tap
  const int k    = ((lane & 16) ? 16 : 0) + j;             // cin
  const int ch   = nt * 16 + (lane & 15);
  dst[tid] = f2bf(w[ch * (CIN * KS) + k * KS + kk]);
}

__global__ void k_pack_tc2(const float* __restrict__ w, bf16_t* __restrict__ dst) {
  const int tid = blockIdx.x * blockDim.x + threadIdx.x;   // 6*4*512 = 12288
  if (tid >= 6 * 4 * TILE) return;
  const int j    = tid & 15;
  const int lane = (tid >> 4) & 31;
  const int nt   = (tid >> 9) & 3;
  const int kk   = tid >> 11;                              // 0..5
  const int ks   = kk >> 1;
  const int c    = (kk & 1) * 32 + ((lane & 16) ? 16 : 0) + j;
  const int ch   = nt * 16 + (lane & 15);
  dst[tid] = f2bf(w[ch * (CH * KS) + c * KS + ks]);
}

__global__ void k_pack_cheb(const float* __restrict__ W, bf16_t* __restrict__ dst) {
  const int tid = blockIdx.x * blockDim.x + threadIdx.x;   // 3*2*4*512 = 12288
  if (tid >= 3 * 2 * 4 * TILE) return;
  const int j    = tid & 15;
  const int lane = (tid >> 4) & 31;
  const int nt   = (tid >> 9) & 3;
  const int kk   = (tid >> 11) & 1;
  const int cj   = tid >> 12;                              // Chebyshev order 0..2
  const int c    = kk * 32 + ((lane & 16) ? 16 : 0) + j;
  const int o    = nt * 16 + (lane & 15);
  dst[tid] = f2bf(W[cj * CH * CH + c * CH + o]);
}

// f32 tensor -> bf16 tensor (same layout)
__global__ void k_cvt(const float* __restrict__ src, bf16_t* __restrict__ dst, size_t n) {
  size_t i = (size_t)blockIdx.x * blockDim.x + threadIdx.x;
  const size_t stride = (size_t)gridDim.x * blockDim.x;
  for (; i < n; i += stride) dst[i] = f2bf(src[i]);
}

// ---------------------------------------------------------------------------
// Gated temporal conv 1: (B,16,N,32)bf16 -> (B,14,N,64) f32 + bf16
// block = 64 rows x 64 ch; 4 waves, wave = 16-ch slice across 4 M-tiles
// B-fragments cached in registers, reused by all 4 M-tiles: 36 WMMA / wave
// ---------------------------------------------------------------------------
__global__ void __launch_bounds__(128) k_tconv1(
    const bf16_t* __restrict__ Xh, const bf16_t* __restrict__ wp,   // wp: [gate][3][4][512]
    const float* __restrict__ b1, const float* __restrict__ b2, const float* __restrict__ b3,
    float* __restrict__ out32, bf16_t* __restrict__ outh)
{
  const int lane = threadIdx.x & 31;
  const int wv   = threadIdx.x >> 5;
  const int m0   = blockIdx.x * (16 * MT);
  const int b    = m0 / (T1D * NN);
  const int t    = (m0 / NN) % T1D;
  const int n0   = m0 & (NN - 1);

  v8f c1[MT], c2[MT], c3[MT];
#pragma unroll
  for (int mt = 0; mt < MT; ++mt) { c1[mt] = {}; c2[mt] = {}; c3[mt] = {}; }

#pragma unroll
  for (int kk = 0; kk < 3; ++kk) {
    const bf16_t* tb = wp + (size_t)(kk * 4 + wv) * TILE;
    const v16bf B1 = load_b_pk(tb,                lane);
    const v16bf B2 = load_b_pk(tb + 3 * 4 * TILE, lane);
    const v16bf B3 = load_b_pk(tb + 6 * 4 * TILE, lane);
    const bf16_t* arow = Xh + ((size_t)(b * TT + (t + kk)) * NN + n0) * CIN;
#pragma unroll
    for (int mt = 0; mt < MT; ++mt) {
      const v16bf A = load_a_bf(arow + (size_t)mt * 16 * CIN, CIN, lane);
      c1[mt] = WMMA_BF16(A, B1, c1[mt]);
      c2[mt] = WMMA_BF16(A, B2, c2[mt]);
      c3[mt] = WMMA_BF16(A, B3, c3[mt]);
    }
  }
  const int ch = wv * 16 + (lane & 15);
  const int mh = (lane & 16) ? 8 : 0;
  const float bb1 = b1[ch], bb2 = b2[ch], bb3 = b3[ch];
#pragma unroll
  for (int mt = 0; mt < MT; ++mt) {
#pragma unroll
    for (int v = 0; v < 8; ++v) {
      const float x2 = c2[mt][v] + bb2;
      const float r  = (c1[mt][v] + bb1) + 1.0f / (1.0f + __expf(-x2)) + (c3[mt][v] + bb3);
      const float o  = fmaxf(r, 0.0f);
      const size_t idx = (size_t)(m0 + mt * 16 + mh + v) * CH + ch;
      out32[idx] = o;
      outh[idx]  = f2bf(o);
    }
  }
}

// ---------------------------------------------------------------------------
// Chebyshev einsum: t2h = relu(Tx0@W0 + Tx1@W1 + Tx2@W2 + bias)  (bf16 out)
// 24 WMMA / wave, B reused across 4 M-tiles
// ---------------------------------------------------------------------------
__global__ void __launch_bounds__(128) k_cheb_mm(
    const bf16_t* __restrict__ tx0, const bf16_t* __restrict__ tx1,
    const bf16_t* __restrict__ tx2,
    const bf16_t* __restrict__ wp,          // [j][2][4][512]
    const float* __restrict__ bias, bf16_t* __restrict__ outh)
{
  const int lane = threadIdx.x & 31;
  const int wv   = threadIdx.x >> 5;
  const int m0   = blockIdx.x * (16 * MT);
  const bf16_t* tx[3] = { tx0, tx1, tx2 };

  v8f acc[MT];
#pragma unroll
  for (int mt = 0; mt < MT; ++mt) acc[mt] = {};

#pragma unroll
  for (int j = 0; j < 3; ++j) {
#pragma unroll
    for (int kk = 0; kk < 2; ++kk) {
      const v16bf B = load_b_pk(wp + (size_t)((j * 2 + kk) * 4 + wv) * TILE, lane);
      const bf16_t* arow = tx[j] + (size_t)m0 * CH + kk * 32;
#pragma unroll
      for (int mt = 0; mt < MT; ++mt) {
        const v16bf A = load_a_bf(arow + (size_t)mt * 16 * CH, CH, lane);
        acc[mt] = WMMA_BF16(A, B, acc[mt]);
      }
    }
  }
  const int ch = wv * 16 + (lane & 15);
  const int mh = (lane & 16) ? 8 : 0;
  const float bb = bias[ch];
#pragma unroll
  for (int mt = 0; mt < MT; ++mt)
#pragma unroll
    for (int v = 0; v < 8; ++v)
      outh[(size_t)(m0 + mt * 16 + mh + v) * CH + ch] = f2bf(fmaxf(acc[mt][v] + bb, 0.0f));
}

// ---------------------------------------------------------------------------
// Gated temporal conv 2: (B,14,N,64)bf16 -> (B,12,N,64) f32
// 72 WMMA / wave, 3 B-fragments cached per k-step
// ---------------------------------------------------------------------------
__global__ void __launch_bounds__(128) k_tconv2(
    const bf16_t* __restrict__ t2h, const bf16_t* __restrict__ wp,  // wp: [gate][6][4][512]
    const float* __restrict__ b1, const float* __restrict__ b2, const float* __restrict__ b3,
    float* __restrict__ out)
{
  const int lane = threadIdx.x & 31;
  const int wv   = threadIdx.x >> 5;
  const int m0   = blockIdx.x * (16 * MT);
  const int b    = m0 / (T2D * NN);
  const int t    = (m0 / NN) % T2D;
  const int n0   = m0 & (NN - 1);

  v8f c1[MT], c2[MT], c3[MT];
#pragma unroll
  for (int mt = 0; mt < MT; ++mt) { c1[mt] = {}; c2[mt] = {}; c3[mt] = {}; }

#pragma unroll
  for (int kk = 0; kk < 6; ++kk) {
    const int ks  = kk >> 1;
    const int cof = (kk & 1) * 32;
    const bf16_t* tb = wp + (size_t)(kk * 4 + wv) * TILE;
    const v16bf B1 = load_b_pk(tb,                 lane);
    const v16bf B2 = load_b_pk(tb + 6 * 4 * TILE,  lane);
    const v16bf B3 = load_b_pk(tb + 12 * 4 * TILE, lane);
    const bf16_t* arow = t2h + ((size_t)(b * T1D + (t + ks)) * NN + n0) * CH + cof;
#pragma unroll
    for (int mt = 0; mt < MT; ++mt) {
      const v16bf A = load_a_bf(arow + (size_t)mt * 16 * CH, CH, lane);
      c1[mt] = WMMA_BF16(A, B1, c1[mt]);
      c2[mt] = WMMA_BF16(A, B2, c2[mt]);
      c3[mt] = WMMA_BF16(A, B3, c3[mt]);
    }
  }
  const int ch = wv * 16 + (lane & 15);
  const int mh = (lane & 16) ? 8 : 0;
  const float bb1 = b1[ch], bb2 = b2[ch], bb3 = b3[ch];
#pragma unroll
  for (int mt = 0; mt < MT; ++mt) {
#pragma unroll
    for (int v = 0; v < 8; ++v) {
      const float x2 = c2[mt][v] + bb2;
      const float r  = (c1[mt][v] + bb1) + 1.0f / (1.0f + __expf(-x2)) + (c3[mt][v] + bb3);
      out[(size_t)(m0 + mt * 16 + mh + v) * COUT + ch] = fmaxf(r, 0.0f);
    }
  }
}

// ---------------------------------------------------------------------------
// Graph normalization + sparse propagation
// ---------------------------------------------------------------------------
__global__ void k_zero(float* __restrict__ p, size_t n) {
  size_t i = (size_t)blockIdx.x * blockDim.x + threadIdx.x;
  const size_t stride = (size_t)gridDim.x * blockDim.x;
  for (; i < n; i += stride) p[i] = 0.0f;
}

__global__ void k_deg(const int* __restrict__ row, const int* __restrict__ col,
                      const float* __restrict__ ew, float* __restrict__ deg) {
  const int e = blockIdx.x * blockDim.x + threadIdx.x;
  if (e >= NE) return;
  const float w = (row[e] == col[e]) ? 0.0f : ew[e];
  atomicAdd(&deg[row[e]], w);
}

__global__ void k_dis(const float* __restrict__ deg, float* __restrict__ dis) {
  const int n = blockIdx.x * blockDim.x + threadIdx.x;
  if (n >= NN) return;
  const float d = deg[n];
  dis[n] = (d > 0.0f) ? rsqrtf(d) : 0.0f;
}

__global__ void k_norm(const int* __restrict__ row, const int* __restrict__ col,
                       const float* __restrict__ ew, const float* __restrict__ dis,
                       float* __restrict__ nrm) {
  const int e = blockIdx.x * blockDim.x + threadIdx.x;
  if (e >= NE) return;
  const float w = (row[e] == col[e]) ? 0.0f : ew[e];
  nrm[e] = -dis[row[e]] * w * dis[col[e]];
}

// one block per edge; feature dim = B*T1D*CH = 1792; f32 gather * norm -> scatter-add
__global__ void __launch_bounds__(256) k_prop(
    const float* __restrict__ src, float* __restrict__ dst,
    const int* __restrict__ row, const int* __restrict__ col,
    const float* __restrict__ nrm)
{
  const int e = blockIdx.x;
  const int r = row[e], c = col[e];
  const float nm = nrm[e];
  for (int f = threadIdx.x; f < NB * T1D * CH; f += blockDim.x) {
    const int bt = f >> 6, ch = f & 63;
    const float v = src[((size_t)bt * NN + c) * CH + ch] * nm;
    atomicAdd(&dst[((size_t)bt * NN + r) * CH + ch], v);
  }
}

// tx2h = bf16(2*agg - tx0)   (Tx2 is only ever a WMMA A-operand)
__global__ void k_combine(const float* __restrict__ agg, const float* __restrict__ tx0,
                          bf16_t* __restrict__ dsth, size_t n) {
  size_t i = (size_t)blockIdx.x * blockDim.x + threadIdx.x;
  const size_t stride = (size_t)gridDim.x * blockDim.x;
  for (; i < n; i += stride) dsth[i] = f2bf(2.0f * agg[i] - tx0[i]);
}

// ---------------------------------------------------------------------------
// BatchNorm over node axis: stats over (B, T2D, COUT) = 1536 values per node
// ---------------------------------------------------------------------------
__global__ void __launch_bounds__(256) k_bn(
    const float* __restrict__ t3,
    const float* __restrict__ gamma, const float* __restrict__ beta,
    float* __restrict__ out)
{
  const int n = blockIdx.x;
  __shared__ float ss[256], sq[256];
  float s = 0.0f, q = 0.0f;
  for (int i = threadIdx.x; i < NB * T2D * COUT; i += 256) {
    const int bt = i >> 6, c = i & 63;
    const float v = t3[((size_t)bt * NN + n) * COUT + c];
    s += v; q += v * v;
  }
  ss[threadIdx.x] = s; sq[threadIdx.x] = q;
  __syncthreads();
  for (int st = 128; st > 0; st >>= 1) {
    if ((int)threadIdx.x < st) {
      ss[threadIdx.x] += ss[threadIdx.x + st];
      sq[threadIdx.x] += sq[threadIdx.x + st];
    }
    __syncthreads();
  }
  const float inv = 1.0f / (float)(NB * T2D * COUT);
  const float mu  = ss[0] * inv;
  const float var = sq[0] * inv - mu * mu;
  const float sc  = rsqrtf(var + 1e-5f) * gamma[n];
  const float sh  = beta[n];
  for (int i = threadIdx.x; i < NB * T2D * COUT; i += 256) {
    const int bt = i >> 6, c = i & 63;
    const size_t idx = ((size_t)bt * NN + n) * COUT + c;
    out[idx] = (t3[idx] - mu) * sc + sh;
  }
}

// ---------------------------------------------------------------------------
// Orchestration
// ---------------------------------------------------------------------------
extern "C" void kernel_launch(void* const* d_in, const int* in_sizes, int n_in,
                              void* d_out, int out_size, void* d_ws, size_t ws_size,
                              hipStream_t stream)
{
  (void)in_sizes; (void)n_in; (void)out_size; (void)ws_size;

  const float* X    = (const float*)d_in[0];
  const int*   ei   = (const int*)  d_in[1];
  const float* ew   = (const float*)d_in[2];
  const float* t1w1 = (const float*)d_in[3];  const float* t1b1 = (const float*)d_in[4];
  const float* t1w2 = (const float*)d_in[5];  const float* t1b2 = (const float*)d_in[6];
  const float* t1w3 = (const float*)d_in[7];  const float* t1b3 = (const float*)d_in[8];
  const float* chw  = (const float*)d_in[9];  const float* chb  = (const float*)d_in[10];
  const float* t2w1 = (const float*)d_in[11]; const float* t2b1 = (const float*)d_in[12];
  const float* t2w2 = (const float*)d_in[13]; const float* t2b2 = (const float*)d_in[14];
  const float* t2w3 = (const float*)d_in[15]; const float* t2b3 = (const float*)d_in[16];
  const float* bng  = (const float*)d_in[17]; const float* bnb  = (const float*)d_in[18];

  const int* row = ei;
  const int* col = ei + NE;

  // ---- workspace layout ----
  const size_t SZ14 = (size_t)NB * T1D * NN * CH;    // 3,670,016
  const size_t SZ12 = (size_t)NB * T2D * NN * COUT;  // 3,145,728
  const size_t SZX  = (size_t)NB * TT  * NN * CIN;   // 2,097,152

  float* fws = (float*)d_ws;
  float* t1  = fws;                  // f32 (prop input / combine)
  float* tx1 = t1  + SZ14;           // f32 (prop #1 accum, prop #2 input)
  float* agg = tx1 + SZ14;           // f32 (prop #2 accum)
  float* t3  = agg + SZ14;           // f32 (BN input)
  float* deg = t3  + SZ12;
  float* dis = deg + NN;
  float* nrm = dis + NN;
  bf16_t* hws  = (bf16_t*)(nrm + NE);
  bf16_t* Xh   = hws;                // bf16 activations
  bf16_t* t1h  = Xh  + SZX;
  bf16_t* tx1h = t1h + SZ14;
  bf16_t* tx2h = tx1h + SZ14;
  bf16_t* t2h  = tx2h + SZ14;
  bf16_t* wp1  = t2h + SZ14;         // packed weights
  bf16_t* wpc  = wp1 + (size_t)3 * 3 * 4 * TILE;
  bf16_t* wp2  = wpc + (size_t)3 * 2 * 4 * TILE;

  // ---- weight packing + X conversion ----
  k_pack_tc1 <<<24, 256, 0, stream>>>(t1w1, wp1);
  k_pack_tc1 <<<24, 256, 0, stream>>>(t1w2, wp1 + 3 * 4 * TILE);
  k_pack_tc1 <<<24, 256, 0, stream>>>(t1w3, wp1 + 6 * 4 * TILE);
  k_pack_cheb<<<48, 256, 0, stream>>>(chw, wpc);
  k_pack_tc2 <<<48, 256, 0, stream>>>(t2w1, wp2);
  k_pack_tc2 <<<48, 256, 0, stream>>>(t2w2, wp2 + 6 * 4 * TILE);
  k_pack_tc2 <<<48, 256, 0, stream>>>(t2w3, wp2 + 12 * 4 * TILE);
  k_cvt      <<<1024, 256, 0, stream>>>(X, Xh, SZX);

  // ---- graph normalization ----
  k_zero<<<16, 256, 0, stream>>>(deg, (size_t)NN);
  k_deg <<<NE / 256, 256, 0, stream>>>(row, col, ew, deg);
  k_dis <<<NN / 256, 256, 0, stream>>>(deg, dis);
  k_norm<<<NE / 256, 256, 0, stream>>>(row, col, ew, dis, nrm);

  // ---- temporal conv 1 (WMMA) -> t1 f32 + t1h bf16 ----
  k_tconv1<<<(NB * T1D * NN) / (16 * MT), 128, 0, stream>>>(
      Xh, wp1, t1b1, t1b2, t1b3, t1, t1h);

  // ---- Chebyshev propagation ----
  k_zero   <<<1024, 256, 0, stream>>>(tx1, SZ14);
  k_prop   <<<NE, 256, 0, stream>>>(t1, tx1, row, col, nrm);
  k_cvt    <<<1024, 256, 0, stream>>>(tx1, tx1h, SZ14);
  k_zero   <<<1024, 256, 0, stream>>>(agg, SZ14);
  k_prop   <<<NE, 256, 0, stream>>>(tx1, agg, row, col, nrm);
  k_combine<<<1024, 256, 0, stream>>>(agg, t1, tx2h, SZ14);

  // ---- Chebyshev channel mix + relu (WMMA) -> t2h bf16 ----
  k_cheb_mm<<<(NB * T1D * NN) / (16 * MT), 128, 0, stream>>>(t1h, tx1h, tx2h, wpc, chb, t2h);

  // ---- temporal conv 2 (WMMA) -> t3 f32 ----
  k_tconv2<<<(NB * T2D * NN) / (16 * MT), 128, 0, stream>>>(
      t2h, wp2, t2b1, t2b2, t2b3, t3);

  // ---- BatchNorm over node axis -> d_out ----
  k_bn<<<NN, 256, 0, stream>>>(t3, bng, bnb, (float*)d_out);
}
